// PointwiseAdaConv_7756710936815
// MI455X (gfx1250) — compile-verified, matrix operand order checked by reading, not run on hardware
//
#include <hip/hip_runtime.h>

typedef __attribute__((ext_vector_type(16))) _Float16 v16h;
typedef __attribute__((ext_vector_type(8)))  _Float16 v8h;
typedef __attribute__((ext_vector_type(8)))  float    v8f;

#define NB    8
#define CIN   512
#define SD    512
#define PP    64
#define NG    8
#define HW    16384   // 128*128
#define EPS   1e-5f
#define SLOPE 0.01f

// ---------------------------------------------------------------------------
// 1) AdaptiveAvgPool2d(1) on style_encoding [8,512,32,32] -> s [8,512]
// ---------------------------------------------------------------------------
__launch_bounds__(256)
__global__ void style_pool_kernel(const float* __restrict__ style, float* __restrict__ s) {
    __shared__ float red[256];
    const int nc  = blockIdx.x;          // n*512 + c
    const int tid = threadIdx.x;
    const float4* p = (const float4*)(style + (size_t)nc * 1024);
    float4 v = p[tid];
    red[tid] = v.x + v.y + v.z + v.w;
    __syncthreads();
    for (int off = 128; off > 0; off >>= 1) {
        if (tid < off) red[tid] += red[tid + off];
        __syncthreads();
    }
    if (tid == 0) s[nc] = red[0] * (1.0f / 1024.0f);
}

// ---------------------------------------------------------------------------
// 2) hk = lrelu(s@W1.T + b1), hb = lrelu(s@Wb1.T + bb1)   (8192 outputs)
// ---------------------------------------------------------------------------
__launch_bounds__(256)
__global__ void mlp_hidden_kernel(const float* __restrict__ s,
                                  const float* __restrict__ W1,  const float* __restrict__ b1,
                                  const float* __restrict__ Wb1, const float* __restrict__ bb1,
                                  float* __restrict__ hk, float* __restrict__ hb) {
    const int i = blockIdx.x * 256 + threadIdx.x;      // 0..8191
    const bool isB = i >= NB * CIN;
    const int j = isB ? i - NB * CIN : i;              // 0..4095
    const int n = j >> 9, c = j & 511;
    const float4* Wr = (const float4*)((isB ? Wb1 : W1) + (size_t)c * SD);
    const float4* sv = (const float4*)(s + (size_t)n * SD);
    float acc = isB ? bb1[c] : b1[c];
    #pragma unroll 4
    for (int d = 0; d < SD / 4; ++d) {
        float4 w = Wr[d], x = sv[d];
        acc += w.x * x.x + w.y * x.y + w.z * x.z + w.w * x.w;
    }
    acc = acc > 0.0f ? acc : acc * SLOPE;
    (isB ? hb : hk)[j] = acc;
}

// ---------------------------------------------------------------------------
// 3) kn[n][j] = hk[n]·W2[j] + b2[j]  (j=0..32767).  One thread per j handles
//    all 8 samples so W2 (67 MB) is read exactly once.
// ---------------------------------------------------------------------------
__launch_bounds__(256)
__global__ void mlp_out_kernel(const float* __restrict__ hk,
                               const float* __restrict__ W2, const float* __restrict__ b2,
                               float* __restrict__ kn) {
    __shared__ float sh[NB * CIN];                     // 16 KB
    for (int i = threadIdx.x; i < NB * CIN; i += 256) sh[i] = hk[i];
    __syncthreads();
    const int j = blockIdx.x * 256 + threadIdx.x;      // 0..32767
    float acc[NB];
    const float bj = b2[j];
    #pragma unroll
    for (int n = 0; n < NB; ++n) acc[n] = bj;
    const float4* Wr = (const float4*)(W2 + (size_t)j * CIN);
    for (int d = 0; d < CIN / 4; ++d) {
        float4 w = Wr[d];
        const int db = d * 4;
        #pragma unroll
        for (int n = 0; n < NB; ++n) {
            const float* h = &sh[n * CIN + db];
            acc[n] += w.x * h[0] + w.y * h[1] + w.z * h[2] + w.w * h[3];
        }
    }
    #pragma unroll
    for (int n = 0; n < NB; ++n) kn[(size_t)n * (NG * PP * PP) + j] = acc[n];
}

// ---------------------------------------------------------------------------
// 4) bias[n][c] = hb[n]·Wb2[c] + bb2[c]   (4096 outputs)
// ---------------------------------------------------------------------------
__launch_bounds__(256)
__global__ void bias_out_kernel(const float* __restrict__ hb,
                                const float* __restrict__ Wb2, const float* __restrict__ bb2,
                                float* __restrict__ bias) {
    const int i = blockIdx.x * 256 + threadIdx.x;      // 0..4095
    const int n = i >> 9, c = i & 511;
    const float4* Wr = (const float4*)(Wb2 + (size_t)c * CIN);
    const float4* hv = (const float4*)(hb + (size_t)n * CIN);
    float acc = bb2[c];
    #pragma unroll 4
    for (int d = 0; d < CIN / 4; ++d) {
        float4 w = Wr[d], h = hv[d];
        acc += w.x * h.x + w.y * h.y + w.z * h.z + w.w * h.w;
    }
    bias[i] = acc;
}

// ---------------------------------------------------------------------------
// 5) instance-norm stats: per (n,c) mean + rsqrt(var+eps) over 16384 pixels
// ---------------------------------------------------------------------------
__launch_bounds__(256)
__global__ void inorm_stats_kernel(const float* __restrict__ pred,
                                   float* __restrict__ mu, float* __restrict__ rsig) {
    __shared__ float rs[256], rq[256];
    const int nc = blockIdx.x, tid = threadIdx.x;
    const float4* p = (const float4*)(pred + (size_t)nc * HW);
    float s = 0.0f, q = 0.0f;
    #pragma unroll 4
    for (int i = 0; i < 16; ++i) {
        float4 v = p[tid + i * 256];
        s += v.x + v.y + v.z + v.w;
        q += v.x * v.x + v.y * v.y + v.z * v.z + v.w * v.w;
    }
    rs[tid] = s; rq[tid] = q;
    __syncthreads();
    for (int off = 128; off > 0; off >>= 1) {
        if (tid < off) { rs[tid] += rs[tid + off]; rq[tid] += rq[tid + off]; }
        __syncthreads();
    }
    if (tid == 0) {
        float m = rs[0] * (1.0f / HW);
        float v = rq[0] * (1.0f / HW) - m * m;
        mu[nc]   = m;
        rsig[nc] = rsqrtf(v + EPS);
    }
}

// ---------------------------------------------------------------------------
// 6) Main grouped 1x1 conv with WMMA + async DMA staging.
//    Normalization folded:  out = (K*rsig) @ x  + (bias - (K*rsig)·mu)
//    The raw f32 B tile (64 rows x 256 cols = 64 KB) is DMA'd into LDS with
//    GLOBAL_LOAD_ASYNC_TO_LDS_B128 (ASYNCcnt) so the A-matrix prep overlaps
//    the copy; then a transpose-convert pass builds the f16 [col][K] tile.
//    Grid: x = column tile (64), y = n*8+g (64). 256 threads = 8 wave32s.
// ---------------------------------------------------------------------------
__launch_bounds__(256)
__global__ void ada_conv_wmma_kernel(const float* __restrict__ pred,
                                     const float* __restrict__ kn,
                                     const float* __restrict__ biasv,
                                     const float* __restrict__ mu,
                                     const float* __restrict__ rsig,
                                     float* __restrict__ out) {
    __shared__ __align__(16) float    ldsRaw[64 * 256];  // raw f32 tile, 64 KB
    __shared__ __align__(16) _Float16 ldsA[64 * 64];     // [o][p]   8 KB
    __shared__ __align__(16) _Float16 ldsB[256 * 64];    // [col][p] 32 KB (transposed)
    __shared__ float sMu[64], sRs[64], sBadj[64];

    const int tid   = threadIdx.x;
    const int ng    = blockIdx.y;                 // n*8 + g
    const int col0  = blockIdx.x * 256;
    const int cbase = ng * 64;                    // == n*512 + g*64

    // ---- kick off the async DMA of the raw B tile first (overlaps A prep) --
    // thread t, iter i: row r = (t>>6) + 4*i, col4 = t&63  (256 % 64 == 0)
    const float* gbase = pred + (size_t)cbase * HW + col0;
    {
        const unsigned ldsRawBase = (unsigned)(uintptr_t)&ldsRaw[0];
        const unsigned moff0 = (unsigned)((tid >> 6) * (HW * 4) + (tid & 63) * 16);
        const unsigned loff0 = ldsRawBase + (unsigned)tid * 16;
        #pragma unroll
        for (int i = 0; i < 16; ++i) {
            unsigned moff = moff0 + (unsigned)i * (4u * HW * 4u);  // +4 rows
            unsigned loff = loff0 + (unsigned)i * 4096u;
            asm volatile("global_load_async_to_lds_b128 %0, %1, %2"
                         :: "v"(loff), "v"(moff), "s"(gbase)
                         : "memory");
        }
        if (blockIdx.x + 1 < gridDim.x)  // hint next column tile
            __builtin_prefetch(gbase + (size_t)(tid & 63) * HW + 256 + (tid >> 6) * 64, 0, 1);
    }

    if (tid < 64) { sMu[tid] = mu[cbase + tid]; sRs[tid] = rsig[cbase + tid]; }
    __syncthreads();

    const float* Kp = kn + (size_t)ng * 4096;     // kernel[n][g], row-major 64x64

    // ---- A' = K * rsig (per column), f16 — runs while the DMA is in flight -
    #pragma unroll
    for (int ii = 0; ii < 16; ++ii) {
        const int i = tid + ii * 256;
        ldsA[i] = (_Float16)(Kp[i] * sRs[i & 63]);
    }
    // folded bias: b'[o] = bias[o] - sum_p K[o][p]*rsig[p]*mu[p]
    if (tid < 64) {
        float adj = 0.0f;
        #pragma unroll 4
        for (int p = 0; p < 64; ++p) adj += Kp[tid * 64 + p] * sRs[p] * sMu[p];
        sBadj[tid] = biasv[cbase + tid] - adj;
    }

    // ---- drain the DMA, then transpose-convert f32 -> f16 [col][K] --------
    asm volatile("s_wait_asynccnt 0x0" ::: "memory");
    __syncthreads();
    {
        _Float16* brow = &ldsB[tid * 64];
        #pragma unroll
        for (int r8 = 0; r8 < 8; ++r8) {
            v8h pk;
            #pragma unroll
            for (int j = 0; j < 8; ++j)
                pk[j] = (_Float16)ldsRaw[(r8 * 8 + j) * 256 + tid];
            *(v8h*)&brow[r8 * 8] = pk;            // ds_store_b128
        }
    }
    __syncthreads();

    // ---- WMMA: wave w -> M-tile = w&3, N-half = w>>2 (8 N-tiles each) ------
    const int wv    = tid >> 5;
    const int lane  = tid & 31;
    const int l16   = lane & 15;
    const int mt    = wv & 3;
    const int nhalf = wv >> 2;
    const int koff  = (lane < 16) ? 0 : 8;        // ISA 16-bit A/B lane split

    union V16 { v16h v; v8h h[2]; };

    // Preload A fragments for both K-steps (reused across 8 N-tiles).
    V16 afrag[2];
    const int arow = mt * 16 + l16;
    #pragma unroll
    for (int ks = 0; ks < 2; ++ks) {
        const _Float16* ap = &ldsA[arow * 64 + ks * 32 + koff];
        afrag[ks].h[0] = *(const v8h*)ap;         // K {base..base+7}
        afrag[ks].h[1] = *(const v8h*)(ap + 16);  // K {base+16..base+23}
    }

    const int rbase = mt * 16 + ((lane < 16) ? 0 : 8);
    #pragma unroll
    for (int i = 0; i < 8; ++i) {
        const int nt   = nhalf * 8 + i;
        const int ccol = nt * 16 + l16;
        v8f acc = {};
        V16 bfrag;
        #pragma unroll
        for (int ks = 0; ks < 2; ++ks) {
            const _Float16* bp = &ldsB[ccol * 64 + ks * 32 + koff];
            bfrag.h[0] = *(const v8h*)bp;
            bfrag.h[1] = *(const v8h*)(bp + 16);
            acc = __builtin_amdgcn_wmma_f32_16x16x32_f16(
                false, afrag[ks].v, false, bfrag.v, (short)0, acc, false, false);
        }
        // C/D layout: VGPR r -> M = r (lanes 0-15) / r+8 (lanes 16-31), N = lane%16
        float* orow = out + (size_t)(cbase + rbase) * HW + col0 + ccol;
        #pragma unroll
        for (int r = 0; r < 8; ++r)
            orow[(size_t)r * HW] = acc[r] + sBadj[rbase + r];
    }
}

// ---------------------------------------------------------------------------
extern "C" void kernel_launch(void* const* d_in, const int* in_sizes, int n_in,
                              void* d_out, int out_size, void* d_ws, size_t ws_size,
                              hipStream_t stream) {
    const float* style = (const float*)d_in[0];
    const float* pred  = (const float*)d_in[1];
    const float* W1    = (const float*)d_in[2];
    const float* b1    = (const float*)d_in[3];
    const float* W2    = (const float*)d_in[4];
    const float* b2    = (const float*)d_in[5];
    const float* Wb1   = (const float*)d_in[6];
    const float* bb1   = (const float*)d_in[7];
    const float* Wb2   = (const float*)d_in[8];
    const float* bb2   = (const float*)d_in[9];
    float* out = (float*)d_out;

    // workspace layout (floats): s 4096 | hk 4096 | hb 4096 | kn 262144 |
    //                            bias 4096 | mu 4096 | rsig 4096   (~1.15 MB)
    float* ws   = (float*)d_ws;
    float* s    = ws;
    float* hk   = s    + 4096;
    float* hb   = hk   + 4096;
    float* kn   = hb   + 4096;
    float* bias = kn   + 262144;
    float* mu   = bias + 4096;
    float* rsig = mu   + 4096;

    style_pool_kernel <<<4096, 256, 0, stream>>>(style, s);
    mlp_hidden_kernel <<<32,   256, 0, stream>>>(s, W1, b1, Wb1, bb1, hk, hb);
    mlp_out_kernel    <<<128,  256, 0, stream>>>(hk, W2, b2, kn);
    bias_out_kernel   <<<16,   256, 0, stream>>>(hb, Wb2, bb2, bias);
    inorm_stats_kernel<<<4096, 256, 0, stream>>>(pred, mu, rsig);
    ada_conv_wmma_kernel<<<dim3(64, 64), 256, 0, stream>>>(pred, kn, bias, mu, rsig, out);
}
